// Windowattention_25228637897186
// MI455X (gfx1250) — matrix operand order repaired and checked by reading, not compiled, not run on hardware
//
#include <hip/hip_runtime.h>
#include <hip/hip_bf16.h>

// ---------------------------------------------------------------------------
// Window attention (Retinexformer IGAB-style) for MI455X / gfx1250.
// All GEMM-shaped work on v_wmma_f32_16x16x32_bf16 (wave32), softmax on VALU.
// GEMM k-loops are software-pipelined (double-buffered A/B fragments).
// ---------------------------------------------------------------------------

typedef __bf16 v16bf __attribute__((ext_vector_type(16)));
typedef float  v8f   __attribute__((ext_vector_type(8)));

union Frag {
    v16bf          v;
    unsigned short h[16];
    unsigned int   d[8];
    uint4          q[2];
};

// Native f32->bf16 (RNE): lowers to v_cvt_pk_bf16_f32 / v_cvt_bf16-class ops.
__device__ __forceinline__ unsigned short cvt_bf16(float f) {
    __bf16 b = (__bf16)f;
    return __builtin_bit_cast(unsigned short, b);
}

__device__ __forceinline__ v8f wmma_bf16(const Frag& a, const Frag& b, v8f c) {
    return __builtin_amdgcn_wmma_f32_16x16x32_bf16(
        /*neg_a=*/false, a.v, /*neg_b=*/false, b.v,
        /*c_mod=*/(short)0, c, /*reuse_a=*/false, /*reuse_b=*/false);
}

// ---------------------------------------------------------------------------
// K0: convert qkv_w (768x256) and proj_w (256x256) f32 -> bf16
// ---------------------------------------------------------------------------
__global__ void cvt_weights(const float* __restrict__ wq,
                            const float* __restrict__ wp,
                            unsigned short* __restrict__ oq,
                            unsigned short* __restrict__ op) {
    int i = blockIdx.x * blockDim.x + threadIdx.x;
    if (i < 768 * 256) oq[i] = cvt_bf16(wq[i]);
    if (i < 256 * 256) op[i] = cvt_bf16(wp[i]);
}

// ---------------------------------------------------------------------------
// K1: QKV GEMM.  out[m,n] = sum_k X[m,k] * Wq[n,k] + bias[n]
//   M = 262144, N = 768, K = 256.  Block = 8 waves, tile 64(M) x 128(N).
//   k-loop double-buffered: stage k+1 loads issue before stage k WMMAs.
//   Epilogue: scale Q, modulate V by illu, scatter bf16 into [b][h][l][32].
// ---------------------------------------------------------------------------
__global__ __launch_bounds__(256) void qkv_gemm(
        const float* __restrict__ X, const float* __restrict__ illu,
        const unsigned short* __restrict__ Wq, const float* __restrict__ bias,
        unsigned short* __restrict__ Qb, unsigned short* __restrict__ Kb,
        unsigned short* __restrict__ Vb) {
    const int t      = threadIdx.x & 31;
    const int wid    = threadIdx.x >> 5;
    const int mwave  = wid & 3, nwave = wid >> 2;
    const int m0     = blockIdx.y * 64  + mwave * 16;   // M tile base
    const int n0     = blockIdx.x * 128 + nwave * 64;   // N tile base
    const int lane16 = t & 15, kh = t >> 4;
    const int arow   = m0 + lane16;                     // A fragment row (lane = M)

    v8f    acc[4] = {};
    float2 af[2][8];
    Frag   B[2][8 / 2];   // [stage][nt]

    // ---- prologue: stage k0 = 0 into buffer 0 ----
    {
        const float* xr = X + arow * 256 + kh * 16;
#pragma unroll
        for (int j = 0; j < 8; ++j)
            af[0][j] = *(const float2*)(xr + j * 2);
#pragma unroll
        for (int nt = 0; nt < 4; ++nt) {
            const uint4* wp = (const uint4*)(Wq + (n0 + nt * 16 + lane16) * 256 + kh * 16);
            B[0][nt].q[0] = wp[0]; B[0][nt].q[1] = wp[1];
        }
    }

#pragma unroll
    for (int ks = 0; ks < 8; ++ks) {          // k0 = ks*32, full unroll folds cur
        const int cur = ks & 1;
        if (ks < 7) {                         // ---- issue stage k+1 loads ----
            const int kn = (ks + 1) * 32;
            const float* xr = X + arow * 256 + kn + kh * 16;
            __builtin_prefetch(xr + 32, 0, 3);
#pragma unroll
            for (int j = 0; j < 8; ++j)
                af[cur ^ 1][j] = *(const float2*)(xr + j * 2);
#pragma unroll
            for (int nt = 0; nt < 4; ++nt) {
                const uint4* wp = (const uint4*)(Wq + (n0 + nt * 16 + lane16) * 256 + kn + kh * 16);
                B[cur ^ 1][nt].q[0] = wp[0]; B[cur ^ 1][nt].q[1] = wp[1];
            }
        }
        // ---- consume stage k ----
        Frag A;
#pragma unroll
        for (int j = 0; j < 8; ++j) {
            A.v[2 * j]     = (__bf16)af[cur][j].x;
            A.v[2 * j + 1] = (__bf16)af[cur][j].y;
        }
#pragma unroll
        for (int nt = 0; nt < 4; ++nt)
            acc[nt] = wmma_bf16(A, B[cur][nt], acc[nt]);
    }

    const float scale = 0.17677669529663687f;   // 32^-0.5
#pragma unroll
    for (int nt = 0; nt < 4; ++nt) {
        int n = n0 + nt * 16 + lane16;
        // n>>8 is uniform across the wave (16-wide tile never crosses 256): scalarize.
        int which = __builtin_amdgcn_readfirstlane(n >> 8);
        int rem   = n & 255;
        int head  = rem >> 5, dcol = rem & 31;
        unsigned short* base = (which == 0) ? Qb : ((which == 1) ? Kb : Vb);
        float bv = bias[n];
#pragma unroll
        for (int r = 0; r < 8; ++r) {
            int   m = m0 + r + kh * 8;          // C/D layout: M = r + 8*(lane>=16)
            float v = acc[nt][r] + bv;
            if (which == 0)      v *= scale;
            else if (which == 2) v *= illu[m * 256 + rem];
            int bw = m >> 6, l = m & 63;
            base[(((bw * 8 + head) * 64) + l) * 32 + dcol] = cvt_bf16(v);
        }
    }
}

// ---------------------------------------------------------------------------
// K2: per-(window, head) attention.  1 wave per head, 4 waves per block.
//   S = Q K^T (d=32 -> single WMMA per 16x16 tile) -> +bias +mask -> softmax
//   -> P (bf16 in LDS) -> P V -> X2[b][l][h*32+d] bf16
// ---------------------------------------------------------------------------
__global__ __launch_bounds__(128) void win_attn(
        const unsigned short* __restrict__ Qb, const unsigned short* __restrict__ Kb,
        const unsigned short* __restrict__ Vb, const float* __restrict__ rpb,
        const int* __restrict__ rel, const float* __restrict__ mask,
        unsigned short* __restrict__ X2) {
    __shared__ float          Ssh[4][16 * 64];
    __shared__ unsigned short Psh[4][16 * 64];

    const int t      = threadIdx.x & 31;
    const int wv     = threadIdx.x >> 5;
    const int b      = blockIdx.x;
    const int head   = blockIdx.y * 4 + wv;
    const int w      = b & 1023;                // mask window index (b % nw)
    const int lane16 = t & 15, kh = t >> 4;

    const unsigned short* Qh = Qb + (size_t)(b * 8 + head) * 64 * 32;
    const unsigned short* Kh = Kb + (size_t)(b * 8 + head) * 64 * 32;
    const unsigned short* Vh = Vb + (size_t)(b * 8 + head) * 64 * 32;

    // K^T as B fragments: B[kdim][j] = K[j][kdim]; lane = j, contiguous kdim.
    Frag KB[4];
#pragma unroll
    for (int jt = 0; jt < 4; ++jt) {
        const uint4* p = (const uint4*)(Kh + (jt * 16 + lane16) * 32 + kh * 16);
        KB[jt].q[0] = p[0]; KB[jt].q[1] = p[1];
    }
    // V as B fragments: B[kl][d]; lane = d column, elements walk kl (stride-32 gather).
    Frag VB[2][2];
#pragma unroll
    for (int nt = 0; nt < 2; ++nt)
#pragma unroll
        for (int kt = 0; kt < 2; ++kt)
#pragma unroll
            for (int e = 0; e < 16; ++e) {
                int kl = kt * 32 + kh * 16 + e;
                VB[nt][kt].h[e] = Vh[kl * 32 + nt * 16 + lane16];
            }

    for (int it = 0; it < 4; ++it) {
        // Q A-fragment for this 16-row block (K = full d = 32)
        Frag A;
        const unsigned short* qr = Qh + (it * 16 + lane16) * 32 + kh * 16;
#pragma unroll
        for (int j = 0; j < 8; ++j) {
            int koff = j * 2;
            A.d[j] = *(const unsigned int*)(qr + koff);
        }
        v8f s[4];
#pragma unroll
        for (int jt = 0; jt < 4; ++jt) {
            v8f z = {};
            s[jt] = wmma_bf16(A, KB[jt], z);
        }
#pragma unroll
        for (int jt = 0; jt < 4; ++jt)
#pragma unroll
            for (int r = 0; r < 8; ++r)
                Ssh[wv][(r + kh * 8) * 64 + jt * 16 + lane16] = s[jt][r];
        __syncthreads();

        // softmax over 64 cols; lanes 0..15 each own one row
        if (t < 16) {
            int il = it * 16 + t;
            float*       Srow = &Ssh[wv][t * 64];
            const float* mrow = mask + ((size_t)w * 64 + il) * 64;
            const int*   rrow = rel + il * 64;
            float mx = -3.0e38f;
#pragma unroll 8
            for (int j = 0; j < 64; ++j) {
                float sv = Srow[j] + rpb[rrow[j] * 8 + head] + mrow[j];
                Srow[j] = sv;
                mx = fmaxf(mx, sv);
            }
            float sum = 0.f;
#pragma unroll 8
            for (int j = 0; j < 64; ++j) {
                float e = __expf(Srow[j] - mx);
                Srow[j] = e;
                sum += e;
            }
            float inv = 1.f / sum;
#pragma unroll 8
            for (int j = 0; j < 64; ++j)
                Psh[wv][t * 64 + j] = cvt_bf16(Srow[j] * inv);
        }
        __syncthreads();

        // out(16x32) = P(16x64) @ V(64x32)
#pragma unroll
        for (int nt = 0; nt < 2; ++nt) {
            v8f o = {};
#pragma unroll
            for (int kt = 0; kt < 2; ++kt) {
                Frag PA;
#pragma unroll
                for (int j = 0; j < 8; ++j) {
                    int K = kt * 32 + kh * 16 + j * 2;
                    PA.d[j] = *(const unsigned int*)&Psh[wv][lane16 * 64 + K];
                }
                o = wmma_bf16(PA, VB[nt][kt], o);
            }
#pragma unroll
            for (int r = 0; r < 8; ++r) {
                int l   = it * 16 + r + kh * 8;
                int col = head * 32 + nt * 16 + lane16;
                X2[((size_t)b * 64 + l) * 256 + col] = cvt_bf16(o[r]);
            }
        }
        __syncthreads();
    }
}

// ---------------------------------------------------------------------------
// K3: output projection.  out[m,n] = sum_k X2[m,k] * Wp[n,k] + bias[n]
//   M = 262144, N = 256, K = 256.  Same pipelined tiling as K1, f32 output.
// ---------------------------------------------------------------------------
__global__ __launch_bounds__(256) void proj_gemm(
        const unsigned short* __restrict__ X2, const unsigned short* __restrict__ Wp,
        const float* __restrict__ bias, float* __restrict__ out) {
    const int t      = threadIdx.x & 31;
    const int wid    = threadIdx.x >> 5;
    const int mwave  = wid & 3, nwave = wid >> 2;
    const int m0     = blockIdx.y * 64  + mwave * 16;
    const int n0     = blockIdx.x * 128 + nwave * 64;
    const int lane16 = t & 15, kh = t >> 4;
    const int arow   = m0 + lane16;

    v8f  acc[4] = {};
    Frag A[2], B[2][4];

    {
        const unsigned short* xr = X2 + (size_t)arow * 256 + kh * 16;
#pragma unroll
        for (int j = 0; j < 8; ++j)
            A[0].d[j] = *(const unsigned int*)(xr + j * 2);
#pragma unroll
        for (int nt = 0; nt < 4; ++nt) {
            const uint4* wp = (const uint4*)(Wp + (n0 + nt * 16 + lane16) * 256 + kh * 16);
            B[0][nt].q[0] = wp[0]; B[0][nt].q[1] = wp[1];
        }
    }

#pragma unroll
    for (int ks = 0; ks < 8; ++ks) {
        const int cur = ks & 1;
        if (ks < 7) {
            const int kn = (ks + 1) * 32;
            const unsigned short* xr = X2 + (size_t)arow * 256 + kn + kh * 16;
            __builtin_prefetch(xr + 32, 0, 3);
#pragma unroll
            for (int j = 0; j < 8; ++j)
                A[cur ^ 1].d[j] = *(const unsigned int*)(xr + j * 2);
#pragma unroll
            for (int nt = 0; nt < 4; ++nt) {
                const uint4* wp = (const uint4*)(Wp + (n0 + nt * 16 + lane16) * 256 + kn + kh * 16);
                B[cur ^ 1][nt].q[0] = wp[0]; B[cur ^ 1][nt].q[1] = wp[1];
            }
        }
#pragma unroll
        for (int nt = 0; nt < 4; ++nt)
            acc[nt] = wmma_bf16(A[cur], B[cur][nt], acc[nt]);
    }

#pragma unroll
    for (int nt = 0; nt < 4; ++nt) {
        int   n  = n0 + nt * 16 + lane16;
        float bv = bias[n];
#pragma unroll
        for (int r = 0; r < 8; ++r) {
            int m = m0 + r + kh * 8;
            out[(size_t)m * 256 + n] = acc[nt][r] + bv;
        }
    }
}

// ---------------------------------------------------------------------------
// Host launch
// ---------------------------------------------------------------------------
extern "C" void kernel_launch(void* const* d_in, const int* in_sizes, int n_in,
                              void* d_out, int out_size, void* d_ws, size_t ws_size,
                              hipStream_t stream) {
    (void)in_sizes; (void)n_in; (void)out_size; (void)ws_size;

    const float* x_in   = (const float*)d_in[0];
    const float* illu   = (const float*)d_in[1];
    const float* mask   = (const float*)d_in[2];
    const float* rpb    = (const float*)d_in[3];
    const float* qkv_w  = (const float*)d_in[4];
    const float* qkv_b  = (const float*)d_in[5];
    const float* proj_w = (const float*)d_in[6];
    const float* proj_b = (const float*)d_in[7];
    const int*   rel    = (const int*)d_in[8];
    float*       out    = (float*)d_out;

    char* ws = (char*)d_ws;
    const size_t QKV_ELT = (size_t)4096 * 8 * 64 * 32;      // per Q/K/V buffer
    unsigned short* Wq = (unsigned short*)(ws);                              // 768*256 bf16
    unsigned short* Wp = (unsigned short*)(ws + 393216);                     // 256*256 bf16
    unsigned short* Qb = (unsigned short*)(ws + 524288);
    unsigned short* Kb = (unsigned short*)(ws + 524288 + 2 * QKV_ELT);
    unsigned short* Vb = (unsigned short*)(ws + 524288 + 4 * QKV_ELT);
    unsigned short* X2 = (unsigned short*)(ws + 524288 + 6 * QKV_ELT);

    cvt_weights<<<768, 256, 0, stream>>>(qkv_w, proj_w, Wq, Wp);
    qkv_gemm<<<dim3(6, 4096), 256, 0, stream>>>(x_in, illu, Wq, qkv_b, Qb, Kb, Vb);
    win_attn<<<dim3(4096, 2), 128, 0, stream>>>(Qb, Kb, Vb, rpb, rel, mask, X2);
    proj_gemm<<<dim3(2, 4096), 256, 0, stream>>>(X2, Wp, proj_b, out);
}
